// MultiHeadALiBi_6133213299480
// MI455X (gfx1250) — matrix operand order repaired and checked by reading, not compiled, not run on hardware
//
#include <hip/hip_runtime.h>
#include <math.h>

// Problem constants
#define BB 4
#define SS 2048
#define EE 512
#define HH 8
#define DD 64

typedef __attribute__((ext_vector_type(16))) _Float16 v16h;
typedef __attribute__((ext_vector_type(8)))  _Float16 v8h;
typedef __attribute__((ext_vector_type(8)))  float    v8f;
typedef __attribute__((ext_vector_type(4)))  unsigned int u32x4;
typedef __attribute__((ext_vector_type(8)))  int          i32x8;
typedef __attribute__((ext_vector_type(4)))  int          i32x4;

#if __has_builtin(__builtin_amdgcn_tensor_load_to_lds)
#define USE_TDM 1
#endif

union F16Frag { v16h v; v8h h[2]; _Float16 e[16]; };

__device__ __forceinline__ v8f wmma16(v16h a, v16h b, v8f c) {
  // D = A(16x32 f16) * B(32x16 f16) + C(16x16 f32)
  return __builtin_amdgcn_wmma_f32_16x16x32_f16(false, a, false, b, (short)0, c,
                                                false, false);
}

__device__ __forceinline__ float rowmax16(float v) {
  v = fmaxf(v, __shfl_xor(v, 1));
  v = fmaxf(v, __shfl_xor(v, 2));
  v = fmaxf(v, __shfl_xor(v, 4));
  v = fmaxf(v, __shfl_xor(v, 8));
  return v;
}
__device__ __forceinline__ float rowsum16(float v) {
  v += __shfl_xor(v, 1);
  v += __shfl_xor(v, 2);
  v += __shfl_xor(v, 4);
  v += __shfl_xor(v, 8);
  return v;
}

#ifdef USE_TDM
// TDM: DMA one 16-row x 32-byte tile (row stride 2048 B) of a byte tensor
// into LDS at lds_off.  D# per CDNA5 ISA 8.3/8.4:
//  g0: [1:0]=count=1, [63:32]=lds_addr, [120:64]=global_addr, [127:126]=type=2
//  g1: data_size=1B, tensor_dim0=2048, tensor_dim1=8192,
//      tile_dim0=32, tile_dim1=16, tensor_dim0_stride=2048
__device__ __forceinline__ void tdm_load_mask_tile(unsigned lds_off,
                                                   const unsigned char* gp) {
  const unsigned long long ga = (unsigned long long)(uintptr_t)gp;
  u32x4 g0 = {1u, lds_off, (unsigned)ga, (unsigned)(ga >> 32) | 0x80000000u};
  i32x8 g1 = {0,
              0x08000000,   // tensor_dim0[15:0]=2048 at bits[63:48]
              0x20000000,   // tensor_dim1[15:0]=8192 at bits[95:80]
              0x00200000,   // tile_dim0=32 at bits[127:112]
              16,           // tile_dim1=16 at bits[143:128]
              2048,         // tensor_dim0_stride[31:0]
              0, 0};
  i32x4 z4 = {0, 0, 0, 0};
#if defined(__clang_major__) && (__clang_major__ >= 23)
  i32x8 z8 = {0, 0, 0, 0, 0, 0, 0, 0};
  __builtin_amdgcn_tensor_load_to_lds(g0, g1, z4, z4, z8, 0);
#else
  __builtin_amdgcn_tensor_load_to_lds(g0, g1, z4, z4, 0);
#endif
}

__device__ __forceinline__ void tdm_wait0() {
#if __has_builtin(__builtin_amdgcn_s_wait_tensorcnt)
  __builtin_amdgcn_s_wait_tensorcnt((short)0);
#else
  asm volatile("s_wait_tensorcnt 0x0" ::: "memory");
#endif
}
#endif  // USE_TDM

// -------------------------------------------------------------------------
// Kernel 1: per-head projections.  z=0: qh = q@Wq^T+bq -> [B,H,S,D] f16
//                                  z=1: kh             -> [B,H,S,D] f16
//                                  z=2: vhT            -> [B,H,D,S] f16 (transposed)
// grid = (M/128, H, 3), block = 256 (8 waves, each wave: 16 M-rows x 64 N)
// -------------------------------------------------------------------------
__global__ __launch_bounds__(256) void proj_kernel(
    const float* __restrict__ q, const float* __restrict__ k,
    const float* __restrict__ v, const float* __restrict__ Wq,
    const float* __restrict__ Wk, const float* __restrict__ Wv,
    const float* __restrict__ bq, const float* __restrict__ bk,
    const float* __restrict__ bv, _Float16* __restrict__ qh,
    _Float16* __restrict__ kh, _Float16* __restrict__ vhT) {
  __shared__ _Float16 Wlds[DD * EE];  // 64 KB (of 320 KB WGP LDS)

  const int tid = threadIdx.x;
  const int w = tid >> 5, lane = tid & 31;
  const int h = blockIdx.y, z = blockIdx.z;
  const int m0 = blockIdx.x * 128;

  const float* X; const float* W; const float* bias;
  if (z == 0)      { X = q; W = Wq; bias = bq; }
  else if (z == 1) { X = k; W = Wk; bias = bk; }
  else             { X = v; W = Wv; bias = bv; }

  // stage W[h] (D x E f32) as f16 in LDS
  const float* Wh = W + (size_t)h * DD * EE;
  for (int i = tid; i < DD * EE; i += 256) Wlds[i] = (_Float16)Wh[i];
  __syncthreads();

  const int rowA = m0 + w * 16 + (lane & 15);   // A-fragment row for this lane
  const int koff = (lane >> 4) * 8;             // A-fragment K sub-offset

  v8f acc[4];
#pragma unroll
  for (int t = 0; t < 4; ++t) acc[t] = {};

  for (int kb = 0; kb < EE; kb += 32) {
    if (kb + 32 < EE)
      __builtin_prefetch(X + (size_t)rowA * EE + kb + 32, 0, 3);
    F16Frag a;
    const float* xp = X + (size_t)rowA * EE + kb + koff;
#pragma unroll
    for (int j = 0; j < 8; ++j) {
      a.e[j]     = (_Float16)xp[j];
      a.e[8 + j] = (_Float16)xp[16 + j];
    }
#pragma unroll
    for (int t = 0; t < 4; ++t) {
      const int n = t * 16 + (lane & 15);
      const _Float16* bp = &Wlds[(size_t)n * EE + kb + (lane >> 4) * 8];
      F16Frag bf;
      bf.h[0] = *(const v8h*)bp;
      bf.h[1] = *(const v8h*)(bp + 16);
      acc[t] = wmma16(a.v, bf.v, acc[t]);
    }
  }

  // store (+bias).  C layout: VGPR r, lanes0-15 -> M=r, lanes16-31 -> M=r+8, N=lane%16
#pragma unroll
  for (int t = 0; t < 4; ++t) {
    const int n = t * 16 + (lane & 15);
    const float bval = bias[h * DD + n];
#pragma unroll
    for (int r = 0; r < 8; ++r) {
      const int gm = m0 + w * 16 + r + ((lane >> 4) << 3);
      const int b = gm >> 11, s = gm & (SS - 1);
      const float val = acc[t][r] + bval;
      if (z == 0)
        qh[(((size_t)(b * HH + h)) * SS + s) * DD + n] = (_Float16)val;
      else if (z == 1)
        kh[(((size_t)(b * HH + h)) * SS + s) * DD + n] = (_Float16)val;
      else
        vhT[(((size_t)(b * HH + h)) * DD + n) * SS + s] = (_Float16)val;
    }
  }
}

// -------------------------------------------------------------------------
// Kernel 2: fused attention.  One block = (batch b, 16 Q rows, ALL 8 heads).
// wave w == head h.  Online softmax (exp term) + independent bias-GEMM term:
//   att = (!attn * P) @ V  -  (!attn * !alibi * dist*bs/rm) @ V
// Mask tiles DMA'd by the Tensor Data Mover into LDS, shared by all 8 heads.
// -------------------------------------------------------------------------
__global__ __launch_bounds__(256) void attn_kernel(
    const _Float16* __restrict__ qh, const _Float16* __restrict__ kh,
    const _Float16* __restrict__ vhT, const float* __restrict__ coords_q,
    const float* __restrict__ coords_k, const unsigned char* __restrict__ attn_m,
    const unsigned char* __restrict__ alibi_m,
    const float* __restrict__ running_mean, const float* __restrict__ bias_scale,
    _Float16* __restrict__ att) {
  __shared__ unsigned char s_attn[16 * 32];
  __shared__ unsigned char s_alibi[16 * 32];
  __shared__ float s_ck[32 * 2];
  __shared__ _Float16 s_P[HH][16 * 32];   // P'  tile per wave (C->A relayout)
  __shared__ _Float16 s_Bw[HH][16 * 32];  // bias tile per wave

  const int tid = threadIdx.x;
  const int h = tid >> 5, lane = tid & 31;
  const int b = blockIdx.x >> 7;              // S/16 = 128 q-tiles per batch
  const int q0 = (blockIdx.x & 127) * 16;

  const size_t bh = (size_t)(b * HH + h);
  const float inv_rm = bias_scale[h] / running_mean[h];

#ifdef USE_TDM
  // LDS aperture puts the LDS byte offset in addr[31:0] of the generic address
  const unsigned lds_attn  = (unsigned)(uintptr_t)(void*)s_attn;
  const unsigned lds_alibi = (unsigned)(uintptr_t)(void*)s_alibi;
#endif

  // Q A-fragments (two K=32 chunks over D=64), loaded once
  F16Frag aq[2];
  {
    const _Float16* qb =
        qh + (bh * SS + q0 + (lane & 15)) * DD + (lane >> 4) * 8;
    aq[0].h[0] = *(const v8h*)(qb);
    aq[0].h[1] = *(const v8h*)(qb + 16);
    aq[1].h[0] = *(const v8h*)(qb + 32);
    aq[1].h[1] = *(const v8h*)(qb + 48);
  }

  // per-lane Q coords, aligned to C-layout rows
  float cqx[8], cqy[8];
#pragma unroll
  for (int r = 0; r < 8; ++r) {
    const int row = q0 + r + ((lane >> 4) << 3);
    cqx[r] = coords_q[((size_t)b * SS + row) * 2 + 0];
    cqy[r] = coords_q[((size_t)b * SS + row) * 2 + 1];
  }

  float m[8], Z[8];
  v8f acc_e[4], acc_b[4];
#pragma unroll
  for (int r = 0; r < 8; ++r) { m[r] = -3.0e38f; Z[r] = 0.0f; }
#pragma unroll
  for (int t = 0; t < 4; ++t) { acc_e[t] = {}; acc_b[t] = {}; }

  for (int kk = 0; kk < SS; kk += 32) {
    __syncthreads();  // previous iteration's LDS reads done
    // ---- stage mask tiles (16x32 B, stride S) + coords_k (32x2 f32)
    if (tid < 64) s_ck[tid] = coords_k[((size_t)b * SS + kk) * 2 + tid];
#ifdef USE_TDM
    if (h == 0) {
      const size_t toff = ((size_t)b * SS + q0) * SS + kk;
      tdm_load_mask_tile(lds_attn, attn_m + toff);
      tdm_load_mask_tile(lds_alibi, alibi_m + toff);
      tdm_wait0();
    }
#else
    {
      const int i = tid * 2;
      const int row = i >> 5, col = i & 31;
      const size_t base = ((size_t)b * SS + q0 + row) * SS + kk + col;
      s_attn[i]      = attn_m[base];
      s_attn[i + 1]  = attn_m[base + 1];
      s_alibi[i]     = alibi_m[base];
      s_alibi[i + 1] = alibi_m[base + 1];
    }
#endif
    __syncthreads();

    if (kk + 32 < SS)
      __builtin_prefetch(kh + (bh * SS + kk + 32 + (lane & 15)) * DD, 0, 3);

    // ---- logits for two 16-wide K subtiles
    float lt[2][8], tmax[8];
#pragma unroll
    for (int r = 0; r < 8; ++r) tmax[r] = -3.0e38f;
#pragma unroll
    for (int sub = 0; sub < 2; ++sub) {
      const int k0 = kk + sub * 16;
      const _Float16* kp =
          kh + (bh * SS + k0 + (lane & 15)) * DD + (lane >> 4) * 8;
      F16Frag bk0, bk1;
      bk0.h[0] = *(const v8h*)(kp);
      bk0.h[1] = *(const v8h*)(kp + 16);
      bk1.h[0] = *(const v8h*)(kp + 32);
      bk1.h[1] = *(const v8h*)(kp + 48);
      v8f cl = {};
      cl = wmma16(aq[0].v, bk0.v, cl);
      cl = wmma16(aq[1].v, bk1.v, cl);
#pragma unroll
      for (int r = 0; r < 8; ++r) {
        const float l = cl[r] * 0.125f;  // 1/sqrt(D)
        lt[sub][r] = l;
        tmax[r] = fmaxf(tmax[r], rowmax16(l));
      }
    }

    // ---- online softmax update (exp term only; bias term is not rescaled)
#pragma unroll
    for (int r = 0; r < 8; ++r) {
      const float nm = fmaxf(m[r], tmax[r]);
      const float sc = __expf(m[r] - nm);
      m[r] = nm;
      Z[r] *= sc;
#pragma unroll
      for (int t = 0; t < 4; ++t) acc_e[t][r] *= sc;
    }

    // ---- build P' and bias tiles in LDS (C-layout -> A-layout relay)
#pragma unroll
    for (int sub = 0; sub < 2; ++sub) {
      const int scol = sub * 16 + (lane & 15);
      const float ckx = s_ck[2 * scol], cky = s_ck[2 * scol + 1];
#pragma unroll
      for (int r = 0; r < 8; ++r) {
        const float e = __expf(lt[sub][r] - m[r]);
        Z[r] += rowsum16(e);  // softmax denominator over ALL k (mask is post-hoc)
        const int srow = r + ((lane >> 4) << 3);
        const unsigned char am = s_attn[srow * 32 + scol];
        const unsigned char al = s_alibi[srow * 32 + scol];
        const float dx = cqx[r] - ckx, dy = cqy[r] - cky;
        const float bw = al ? 0.0f : __fsqrt_rn(dx * dx + dy * dy) * inv_rm;
        s_P[h][srow * 32 + scol]  = (_Float16)(am ? 0.0f : e);
        s_Bw[h][srow * 32 + scol] = (_Float16)(am ? 0.0f : bw);
      }
    }
    __syncthreads();

    // ---- PV: two K=32 GEMM-accumulates (exp term and bias term)
    F16Frag ap, ab;
    {
      const _Float16* pp = &s_P[h][(lane & 15) * 32 + (lane >> 4) * 8];
      ap.h[0] = *(const v8h*)pp;
      ap.h[1] = *(const v8h*)(pp + 16);
      const _Float16* bp = &s_Bw[h][(lane & 15) * 32 + (lane >> 4) * 8];
      ab.h[0] = *(const v8h*)bp;
      ab.h[1] = *(const v8h*)(bp + 16);
    }
#pragma unroll
    for (int dt = 0; dt < 4; ++dt) {
      const _Float16* vp =
          vhT + (bh * DD + dt * 16 + (lane & 15)) * SS + kk + (lane >> 4) * 8;
      F16Frag bv;
      bv.h[0] = *(const v8h*)vp;
      bv.h[1] = *(const v8h*)(vp + 16);
      acc_e[dt] = wmma16(ap.v, bv.v, acc_e[dt]);
      acc_b[dt] = wmma16(ab.v, bv.v, acc_b[dt]);
    }
  }

  // ---- epilogue: att = acc_e / Z - acc_b, heads concatenated -> [B,S,E] f16
#pragma unroll
  for (int dt = 0; dt < 4; ++dt) {
#pragma unroll
    for (int r = 0; r < 8; ++r) {
      const float val = acc_e[dt][r] / Z[r] - acc_b[dt][r];
      const int srow = q0 + r + ((lane >> 4) << 3);
      att[((size_t)b * SS + srow) * EE + h * DD + dt * 16 + (lane & 15)] =
          (_Float16)val;
    }
  }
}

// -------------------------------------------------------------------------
// Kernel 3: output projection  out = att @ Wfc^T + bfc   (f32 output)
// grid = (M/128, E/64); block 256 (8 waves x 16 M-rows x 64 N)
// -------------------------------------------------------------------------
__global__ __launch_bounds__(256) void fc_kernel(
    const _Float16* __restrict__ att, const float* __restrict__ Wfc,
    const float* __restrict__ bfc, float* __restrict__ out) {
  __shared__ _Float16 Wlds[64 * EE];  // 64 KB tile: Wfc rows n0..n0+63 as f16

  const int tid = threadIdx.x;
  const int w = tid >> 5, lane = tid & 31;
  const int m0 = blockIdx.x * 128;
  const int n0 = blockIdx.y * 64;

  const float* Wt = Wfc + (size_t)n0 * EE;
  for (int i = tid; i < 64 * EE; i += 256) Wlds[i] = (_Float16)Wt[i];
  __syncthreads();

  const int rowA = m0 + w * 16 + (lane & 15);
  const int koff = (lane >> 4) * 8;

  v8f acc[4];
#pragma unroll
  for (int t = 0; t < 4; ++t) acc[t] = {};

  for (int kb = 0; kb < EE; kb += 32) {
    const _Float16* xp = att + (size_t)rowA * EE + kb + koff;
    F16Frag a;
    a.h[0] = *(const v8h*)xp;
    a.h[1] = *(const v8h*)(xp + 16);
#pragma unroll
    for (int t = 0; t < 4; ++t) {
      const int n = t * 16 + (lane & 15);
      const _Float16* bp = &Wlds[(size_t)n * EE + kb + (lane >> 4) * 8];
      F16Frag bf;
      bf.h[0] = *(const v8h*)bp;
      bf.h[1] = *(const v8h*)(bp + 16);
      acc[t] = wmma16(a.v, bf.v, acc[t]);
    }
  }

#pragma unroll
  for (int t = 0; t < 4; ++t) {
    const int n = n0 + t * 16 + (lane & 15);
    const float bval = bfc[n];
#pragma unroll
    for (int r = 0; r < 8; ++r) {
      const int gm = m0 + w * 16 + r + ((lane >> 4) << 3);
      out[(size_t)gm * EE + n] = acc[t][r] + bval;
    }
  }
}

// -------------------------------------------------------------------------
extern "C" void kernel_launch(void* const* d_in, const int* in_sizes, int n_in,
                              void* d_out, int out_size, void* d_ws,
                              size_t ws_size, hipStream_t stream) {
  (void)in_sizes; (void)n_in; (void)out_size; (void)ws_size;

  const float* q        = (const float*)d_in[0];
  const float* k        = (const float*)d_in[1];
  const float* v        = (const float*)d_in[2];
  const float* coords_q = (const float*)d_in[3];
  const float* coords_k = (const float*)d_in[4];
  const unsigned char* attn_m  = (const unsigned char*)d_in[5];
  const unsigned char* alibi_m = (const unsigned char*)d_in[6];
  const float* Wq  = (const float*)d_in[7];
  const float* bq  = (const float*)d_in[8];
  const float* Wk  = (const float*)d_in[9];
  const float* bk  = (const float*)d_in[10];
  const float* Wv  = (const float*)d_in[11];
  const float* bv  = (const float*)d_in[12];
  const float* rm  = (const float*)d_in[13];
  const float* bs  = (const float*)d_in[14];
  const float* Wfc = (const float*)d_in[15];
  const float* bfc = (const float*)d_in[16];

  // workspace: qh | kh | vhT | att  (all f16)
  const size_t n_head = (size_t)BB * HH * SS * DD;  // 4 Mi elems
  _Float16* qh  = (_Float16*)d_ws;
  _Float16* kh  = qh + n_head;
  _Float16* vhT = kh + n_head;
  _Float16* att = vhT + n_head;

  proj_kernel<<<dim3((BB * SS) / 128, HH, 3), 256, 0, stream>>>(
      q, k, v, Wq, Wk, Wv, bq, bk, bv, qh, kh, vhT);

  attn_kernel<<<dim3(BB * (SS / 16)), 256, 0, stream>>>(
      qh, kh, vhT, coords_q, coords_k, attn_m, alibi_m, rm, bs, att);

  fc_kernel<<<dim3((BB * SS) / 128, EE / 64), 256, 0, stream>>>(
      att, Wfc, bfc, (float*)d_out);
}
